// UniModalAttention_3083786518818
// MI455X (gfx1250) — matrix operand order repaired and checked by reading, not compiled, not run on hardware
//
#include <hip/hip_runtime.h>
#include <hip/hip_bf16.h>
#include <math.h>

// ---------------------------------------------------------------------------
// UniModalAttention for MI455X (gfx1250, wave32, WMMA 16x16x32 f16).
//
// Shapes: N=8 H=8 T=L=1024 D=128.
//   k0: transpose+cvt weights to f16 (WMMA B-operand friendly layout)
//   k1: K = relu(X@Wk1+bk1)@Wk2+bk2 -> row-major f16 [token][c]
//       V likewise                  -> TRANSPOSED f16 [nh][c][l]
//       (C/D fragment layout has consecutive token indices per lane, so the
//        transposed V store is a single 16B store per tile - no LDS shuffle)
//   k2: S = Q@K^T ; P = mask ? exp(-S^2/tau) : 0 ; den += mask ? e : 1
//       O = (P@V) / den   (single pass over L: logits <= 0, exp never
//       overflows, so no flash-attention rescaling is needed)
// Workspace: 4*128*128 (wt) + 2*65536*128 (K,V) f16 = ~33.7 MB.
// ---------------------------------------------------------------------------

typedef _Float16 v16h __attribute__((ext_vector_type(16)));
typedef _Float16 h8   __attribute__((ext_vector_type(8)));
typedef float    v8f  __attribute__((ext_vector_type(8)));
typedef float    f4   __attribute__((ext_vector_type(4)));

__device__ __forceinline__ v8f wmma16(v16h a, v16h b, v8f c) {
  // D = A(16x32 f16) x B(32x16 f16) + C(16x16 f32)
  return __builtin_amdgcn_wmma_f32_16x16x32_f16(
      /*neg_a=*/false, a, /*neg_b=*/false, b,
      /*c_mod=*/(short)0, c, /*reuse_a=*/false, /*reuse_b=*/false);
}

__device__ __forceinline__ v16h combine16(h8 lo, h8 hi) {
  union { v16h v; h8 h[2]; } u;
  u.h[0] = lo; u.h[1] = hi;
  return u.v;
}

// Load one WMMA A-operand chunk from an f32 row.
// A layout (contiguous, matching B): lanes m / m+16 both hold row m; lane<16 holds
// K={0..15}, lane>=16 holds K={16..31}. Caller passes
// p = row_base + kc*32 + hi*16 so the 16 elements are p[0..15].
__device__ __forceinline__ v16h load_a_f32(const float* p) {
  f4 a0 = *(const f4*)(p);
  f4 a1 = *(const f4*)(p + 4);
  f4 b0 = *(const f4*)(p + 8);
  f4 b1 = *(const f4*)(p + 12);
  v16h r;
#pragma unroll
  for (int i = 0; i < 4; ++i) {
    r[i]      = (_Float16)a0[i];
    r[4 + i]  = (_Float16)a1[i];
    r[8 + i]  = (_Float16)b0[i];
    r[12 + i] = (_Float16)b1[i];
  }
  return r;
}

// ---------------------------------------------------------------------------
// k0: wt[w][o*128+c] = W_w[c*128+o] as f16  (w = Wk1,Wk2,Wv1,Wv2)
// ---------------------------------------------------------------------------
__global__ void wprep_kernel(const float* __restrict__ Wk1,
                             const float* __restrict__ Wk2,
                             const float* __restrict__ Wv1,
                             const float* __restrict__ Wv2,
                             _Float16* __restrict__ wt) {
  int i = blockIdx.x * 256 + threadIdx.x;   // 0..16383
  int o = i >> 7, c = i & 127;
  int src = c * 128 + o;
  wt[i]             = (_Float16)Wk1[src];
  wt[16384 + i]     = (_Float16)Wk2[src];
  wt[32768 + i]     = (_Float16)Wv1[src];
  wt[49152 + i]     = (_Float16)Wv2[src];
}

// ---------------------------------------------------------------------------
// k1: KV MLP. 4 waves/block, 16 tokens/wave, 64 tokens/block, 1024 blocks.
// Token index is flat over (n,h,l). K stored [token][c]; V stored [nh][c][l].
// ---------------------------------------------------------------------------
__global__ __launch_bounds__(128) void kv_mlp_kernel(
    const float* __restrict__ X, const _Float16* __restrict__ wt,
    const float* __restrict__ bk1, const float* __restrict__ bk2,
    const float* __restrict__ bv1, const float* __restrict__ bv2,
    _Float16* __restrict__ Kf, _Float16* __restrict__ Vtw) {
  __shared__ __align__(32) _Float16 ldsH[4][16 * 128];  // per-wave hidden tile

  const int tid = threadIdx.x;
  const int wave = tid >> 5, lane = tid & 31;
  const int lo16 = lane & 15, hi = lane >> 4;
  const int r0 = blockIdx.x * 64 + wave * 16;  // first token of this wave
  const int nh = r0 >> 10;                     // blocks never straddle nh
  const int lb = r0 & 1023;                    // l within (n,h)
  _Float16* H = ldsH[wave];

  // X tile as WMMA A-operands (4 chunks of K=32 over D_IN=128)
  v16h xa[4];
#pragma unroll
  for (int kc = 0; kc < 4; ++kc)
    xa[kc] = load_a_f32(X + (size_t)(r0 + lo16) * 128 + kc * 32 + hi * 16);

  for (int w = 0; w < 2; ++w) {  // w=0: K path, w=1: V path
    const _Float16* w1 = wt + (size_t)(w * 2 + 0) * 16384;
    const _Float16* w2 = wt + (size_t)(w * 2 + 1) * 16384;
    const float* b1 = w ? bv1 : bk1;
    const float* b2 = w ? bv2 : bk2;

    // Layer 1: hidden = relu(X @ W1 + b1) -> LDS row-major [16][128]
#pragma unroll
    for (int nt = 0; nt < 8; ++nt) {
      v8f acc = {};
#pragma unroll
      for (int kc = 0; kc < 4; ++kc) {
        // B-operand: lane holds 16 contiguous contraction values of column n
        v16h b = *(const v16h*)(w1 + (size_t)(nt * 16 + lo16) * 128 + kc * 32 + hi * 16);
        acc = wmma16(xa[kc], b, acc);
      }
      float bb = b1[nt * 16 + lo16];
#pragma unroll
      for (int v = 0; v < 8; ++v) {
        float x = acc[v] + bb;
        x = x > 0.f ? x : 0.f;
        H[(v + 8 * hi) * 128 + nt * 16 + lo16] = (_Float16)x;  // C/D layout store
      }
    }

    // Reload hidden as A-operands (layout shuffle through LDS)
    v16h ha[4];
#pragma unroll
    for (int kc = 0; kc < 4; ++kc) {
      h8 p0 = *(const h8*)(H + lo16 * 128 + kc * 32 + hi * 16);
      h8 p1 = *(const h8*)(H + lo16 * 128 + kc * 32 + hi * 16 + 8);
      ha[kc] = combine16(p0, p1);
    }

    // Layer 2: out = hidden @ W2 + b2
#pragma unroll
    for (int nt = 0; nt < 8; ++nt) {
      v8f acc = {};
#pragma unroll
      for (int kc = 0; kc < 4; ++kc) {
        v16h b = *(const v16h*)(w2 + (size_t)(nt * 16 + lo16) * 128 + kc * 32 + hi * 16);
        acc = wmma16(ha[kc], b, acc);
      }
      float bb = b2[nt * 16 + lo16];
      if (w == 0) {
        // K: route through LDS to get row-major [token][c], coalesced store
#pragma unroll
        for (int v = 0; v < 8; ++v)
          H[(v + 8 * hi) * 128 + nt * 16 + lo16] = (_Float16)(acc[v] + bb);
      } else {
        // V: C/D fragment is (c = nt*16+lo16, l = lb + 8*hi + v) -> the 8
        // accumulator elements are CONSECUTIVE l's: one direct 16B store
        // into the transposed [nh][c][l] layout. No LDS shuffle needed.
        h8 hv;
#pragma unroll
        for (int v = 0; v < 8; ++v) hv[v] = (_Float16)(acc[v] + bb);
        *(h8*)(Vtw + (size_t)nh * 131072 + (size_t)(nt * 16 + lo16) * 1024 +
               lb + 8 * hi) = hv;
      }
    }
    if (w == 0) {
#pragma unroll
      for (int j = 0; j < 8; ++j)
        *(h8*)(Kf + (size_t)r0 * 128 + lane * 64 + j * 8) =
            *(const h8*)(H + lane * 64 + j * 8);
    }
  }
}

// ---------------------------------------------------------------------------
// k2: attention. 1 block = one (n,h) x 64 query rows (4 waves x 16 rows).
// Streams L in chunks of 32: K tile row-major LDS, V tile already transposed
// in workspace so the LDS fill is a plain coalesced copy.
// ---------------------------------------------------------------------------
__global__ __launch_bounds__(128) void attn_kernel(
    const float* __restrict__ X, const float* __restrict__ t1,
    const float* __restrict__ Q, const _Float16* __restrict__ Kf,
    const _Float16* __restrict__ Vtw, const float* __restrict__ log_tau,
    float* __restrict__ out) {
  __shared__ __align__(32) _Float16 Kt[32 * 128];   // [l_local][c]
  __shared__ __align__(32) _Float16 Vt[128 * 32];   // [c][l_local]
  __shared__ __align__(32) _Float16 Pt[4][16 * 32]; // per-wave P tile
  __shared__ float t2s[1024];                        // t2[l] for this n

  const int tid = threadIdx.x;
  const int wave = tid >> 5, lane = tid & 31;
  const int lo16 = lane & 15, hi = lane >> 4;
  const int nh = blockIdx.x >> 4;        // 0..63  (n*8+h)
  const int tb = blockIdx.x & 15;
  const int n = nh >> 3;
  const int t0 = tb * 64 + wave * 16;    // first query row of this wave
  const float inv_tau = __expf(-log_tau[0]);

  // t2[l] = X[n,0,l,127]: gather once per block instead of every iteration
  const float* t2base = X + (size_t)n * 8 * 1024 * 128 + 127;
  for (int i = tid; i < 1024; i += 128) t2s[i] = t2base[(size_t)i * 128];

  // Q tile as A-operands
  v16h qa[4];
#pragma unroll
  for (int kc = 0; kc < 4; ++kc)
    qa[kc] = load_a_f32(Q + ((size_t)nh * 1024 + t0 + lo16) * 128 + kc * 32 + hi * 16);

  // t1 values for this lane's 8 rows (row = t0 + v + 8*hi, C/D layout)
  float t1v[8];
#pragma unroll
  for (int v = 0; v < 8; ++v) t1v[v] = t1[n * 1024 + t0 + v + 8 * hi];

  v8f o[8];
#pragma unroll
  for (int ct = 0; ct < 8; ++ct) o[ct] = (v8f){};
  float den[8] = {0.f, 0.f, 0.f, 0.f, 0.f, 0.f, 0.f, 0.f};

  const _Float16* Krow = Kf + (size_t)nh * 1024 * 128;   // [l][c]
  const _Float16* Vrow = Vtw + (size_t)nh * 131072;      // [c][l]

  for (int it = 0; it < 32; ++it) {
    const int l0 = it * 32;
    {  // cooperative tile fill: 128 threads, 16B chunks, fully coalesced
      int r = tid >> 2, q4 = (tid & 3) * 32;
      const h8* ks = (const h8*)(Krow + (size_t)(l0 + r) * 128 + q4);
      h8* kd = (h8*)(Kt + r * 128 + q4);
      kd[0] = ks[0]; kd[1] = ks[1]; kd[2] = ks[2]; kd[3] = ks[3];
      int c = tid >> 2, j8 = (tid & 3) * 8;
      *(h8*)(Vt + c * 32 + j8) = *(const h8*)(Vrow + (size_t)c * 1024 + l0 + j8);
      if (it + 1 < 32) {  // prefetch next tiles (global_prefetch_b8)
        __builtin_prefetch((const void*)(Krow + (size_t)(l0 + 32 + r) * 128 + q4), 0, 1);
        __builtin_prefetch((const void*)(Vrow + (size_t)c * 1024 + l0 + 32 + j8), 0, 1);
      }
    }
    __syncthreads();

    _Float16* P = Pt[wave];
#pragma unroll
    for (int lt = 0; lt < 2; ++lt) {  // two 16-wide S tiles per 32-chunk
      v8f s = {};
#pragma unroll
      for (int kc = 0; kc < 4; ++kc) {
        // B = K^T: lane(lo16)=column l_local, 16 contiguous channel values
        v16h b = *(const v16h*)(Kt + (lt * 16 + lo16) * 128 + kc * 32 + hi * 16);
        s = wmma16(qa[kc], b, s);
      }
      float t2v = t2s[l0 + lt * 16 + lo16];
#pragma unroll
      for (int v = 0; v < 8; ++v) {
        float sv = s[v];
        float e = __expf(-(sv * sv) * inv_tau);  // logits <= 0: safe
        bool mk = t1v[v] >= t2v;
        den[v] += mk ? e : 1.0f;  // masked logit 0 -> exp(0)=1 in denominator
        P[(v + 8 * hi) * 32 + lt * 16 + lo16] = (_Float16)(mk ? e : 0.0f);
      }
    }
    // P: C/D layout -> A layout via wave-private LDS
    h8 p0 = *(const h8*)(P + lo16 * 32 + hi * 16);
    h8 p1 = *(const h8*)(P + lo16 * 32 + hi * 16 + 8);
    v16h pa = combine16(p0, p1);
#pragma unroll
    for (int ct = 0; ct < 8; ++ct) {
      // B = V tile: lane(lo16)=output channel column, contiguous in Vt
      v16h vb = *(const v16h*)(Vt + (ct * 16 + lo16) * 32 + hi * 16);
      o[ct] = wmma16(pa, vb, o[ct]);
    }
    __syncthreads();  // protect Kt/Vt before next chunk overwrites
  }

  // Row denominators: reduce across the 16 lanes of each half-wave group
#pragma unroll
  for (int v = 0; v < 8; ++v) {
    float d = den[v];
    d += __shfl_xor(d, 1);
    d += __shfl_xor(d, 2);
    d += __shfl_xor(d, 4);
    d += __shfl_xor(d, 8);
    den[v] = 1.0f / d;
  }
  float* op = out + ((size_t)nh * 1024 + t0) * 128;
#pragma unroll
  for (int ct = 0; ct < 8; ++ct)
#pragma unroll
    for (int v = 0; v < 8; ++v)
      op[(size_t)(v + 8 * hi) * 128 + ct * 16 + lo16] = o[ct][v] * den[v];
}

// ---------------------------------------------------------------------------
extern "C" void kernel_launch(void* const* d_in, const int* in_sizes, int n_in,
                              void* d_out, int out_size, void* d_ws, size_t ws_size,
                              hipStream_t stream) {
  (void)in_sizes; (void)n_in; (void)out_size; (void)ws_size;
  const float* X       = (const float*)d_in[0];
  const float* t1      = (const float*)d_in[1];
  const float* Q       = (const float*)d_in[2];
  const float* Wk1     = (const float*)d_in[3];
  const float* bk1     = (const float*)d_in[4];
  const float* Wk2     = (const float*)d_in[5];
  const float* bk2     = (const float*)d_in[6];
  const float* Wv1     = (const float*)d_in[7];
  const float* bv1     = (const float*)d_in[8];
  const float* Wv2     = (const float*)d_in[9];
  const float* bv2     = (const float*)d_in[10];
  const float* log_tau = (const float*)d_in[11];
  float* out = (float*)d_out;

  _Float16* wt  = (_Float16*)d_ws;                    // 4 x 128x128
  _Float16* Kf  = wt + (size_t)4 * 16384;             // [token][c]
  _Float16* Vtw = Kf + (size_t)65536 * 128;           // [nh][c][l]

  wprep_kernel<<<64, 256, 0, stream>>>(Wk1, Wk2, Wv1, Wv2, wt);
  kv_mlp_kernel<<<1024, 128, 0, stream>>>(X, wt, bk1, bk2, bv1, bv2, Kf, Vtw);
  attn_kernel<<<1024, 128, 0, stream>>>(X, t1, Q, Kf, Vtw, log_tau, out);
}
